// MultiBoxLoss_89558658056402
// MI455X (gfx1250) — compile-verified, matrix operand order checked by reading, not run on hardware
//
#include <hip/hip_runtime.h>

#define B_ 64
#define M_ 50
#define P_ 24564
#define THRESH 0.2f
#define RATIO 3

typedef __attribute__((ext_vector_type(2)))  float    v2f;
typedef __attribute__((ext_vector_type(8)))  float    v8f;
typedef __attribute__((ext_vector_type(16))) _Float16 v16h;

// ---------------------------------------------------------------------------
// Wave-level (32 lane) sum reduction on the WMMA pipe.
// A-matrix (16x4 f32) layout per CDNA5 ISA: lanes 0-15 hold (M=lane, K=0/1) in
// VGPR0/1, lanes 16-31 hold (M=lane-16, K=2/3). Put lane value in VGPR0, zero
// in VGPR1 -> row M sums to v[M] + v[M+16]. Multiply by all-ones B(4x16):
// D[M][N] = rowsum_M. D VGPR r: lanes 0-15 carry M=r, lanes 16-31 carry M=r+8.
// Sum d[0..7] then shfl_xor(16) -> full wave sum in every lane.
// Requires EXEC == all ones (call sites guarantee this).
// ---------------------------------------------------------------------------
__device__ __forceinline__ float wave_sum_wmma(float v) {
#if __has_builtin(__builtin_amdgcn_wmma_f32_16x16x4_f32)
  v2f a; a[0] = v;    a[1] = 0.0f;
  v2f b; b[0] = 1.0f; b[1] = 1.0f;
  v8f c = {};
  v8f d = __builtin_amdgcn_wmma_f32_16x16x4_f32(false, a, false, b, (short)0,
                                                c, false, false);
#else
  // Fallback: codegen-confirmed f16 WMMA. Lane value lands in a distinct (M,K)
  // (lanes 0-15 -> K=0, lanes 16-31 -> K=8 of row lane%16); ones-B row-sums it.
  v16h a;
#pragma unroll
  for (int i = 0; i < 16; ++i) a[i] = (_Float16)0.0f;
  a[0] = (_Float16)v;
  v16h b;
#pragma unroll
  for (int i = 0; i < 16; ++i) b[i] = (_Float16)1.0f;
  v8f c = {};
  v8f d = __builtin_amdgcn_wmma_f32_16x16x32_f16(false, a, false, b, (short)0,
                                                 c, false, false);
#endif
  float s = d[0] + d[1] + d[2] + d[3] + d[4] + d[5] + d[6] + d[7];
  s += __shfl_xor(s, 16, 32);
  return s;
}

__device__ __forceinline__ float iou_xy(float ax1, float ay1, float ax2, float ay2,
                                        float px1, float py1, float px2, float py2) {
  float ix = fminf(ax2, px2) - fmaxf(ax1, px1);
  float iy = fminf(ay2, py2) - fmaxf(ay1, py1);
  ix = fmaxf(ix, 0.0f);
  iy = fmaxf(iy, 0.0f);
  float inter = ix * iy;
  float aa = (ax2 - ax1) * (ay2 - ay1);
  float ab = (px2 - px1) * (py2 - py1);
  return inter / (aa + ab - inter);
}

// K0: zero accumulators ------------------------------------------------------
__global__ void k0_zero(int* __restrict__ npos, float* __restrict__ acc) {
  int t = threadIdx.x;
  if (t < B_) npos[t] = 0;
  if (t < 3) acc[t] = 0.0f;
}

// K1: per (b,p) best object (max + first-argmax over M) ----------------------
__global__ void k1_best_obj(const float4* __restrict__ boxes4,
                            const float4* __restrict__ priors4,
                            float* __restrict__ overlap,
                            int* __restrict__ objbest) {
  int idx = blockIdx.x * 256 + threadIdx.x;   // exact: B*P % 256 == 0
  int b = idx / P_;
  int p = idx - b * P_;
  float4 pr = priors4[p];
  float px1 = pr.x - pr.z * 0.5f, py1 = pr.y - pr.w * 0.5f;
  float px2 = pr.x + pr.z * 0.5f, py2 = pr.y + pr.w * 0.5f;
  float best = -1.0f;
  int bi = 0;
  for (int m = 0; m < M_; ++m) {
    float4 bx = boxes4[b * M_ + m];
    float o = iou_xy(bx.x, bx.y, bx.z, bx.w, px1, py1, px2, py2);
    if (o > best) { best = o; bi = m; }       // strict ">" => first max
  }
  overlap[idx] = best;
  objbest[idx] = bi;
}

// K2: per (b,m) best prior (first-argmax over P) -----------------------------
__global__ void k2_best_prior(const float4* __restrict__ boxes4,
                              const float4* __restrict__ priors4,
                              int* __restrict__ priorfo) {
  int bm = blockIdx.x;
  int tid = threadIdx.x;
  float4 bx = boxes4[bm];
  float best = -1.0f;
  int bi = 0;
  for (int p = tid; p < P_; p += 256) {
    float4 pr = priors4[p];
    float px1 = pr.x - pr.z * 0.5f, py1 = pr.y - pr.w * 0.5f;
    float px2 = pr.x + pr.z * 0.5f, py2 = pr.y + pr.w * 0.5f;
    float o = iou_xy(bx.x, bx.y, bx.z, bx.w, px1, py1, px2, py2);
    if (o > best) { best = o; bi = p; }
  }
  __shared__ float sv[256];
  __shared__ int   si[256];
  sv[tid] = best; si[tid] = bi;
  __syncthreads();
  for (int s = 128; s > 0; s >>= 1) {
    if (tid < s) {
      float v2 = sv[tid + s]; int i2 = si[tid + s];
      if (v2 > sv[tid] || (v2 == sv[tid] && i2 < si[tid])) { sv[tid] = v2; si[tid] = i2; }
    }
    __syncthreads();
  }
  if (tid == 0) priorfo[bm] = si[0];
}

// K3: force-assign best prior per object (sequential -> last write wins) -----
__global__ void k3_force(const int* __restrict__ priorfo,
                         float* __restrict__ overlap,
                         int* __restrict__ objbest) {
  if (threadIdx.x != 0) return;
  int b = blockIdx.x;
  for (int m = 0; m < M_; ++m) {
    int j = priorfo[b * M_ + m];
    objbest[b * P_ + j] = m;
    overlap[b * P_ + j] = 1.0f;
  }
}

// K4: per-prior loc-L1 + cross-entropy; WMMA wave reductions -----------------
__global__ void k4_losses(const float4* __restrict__ locs4,
                          const float2* __restrict__ scores2,
                          const float4* __restrict__ boxes4,
                          const float4* __restrict__ priors4,
                          const float* __restrict__ overlap,
                          const int* __restrict__ objbest,
                          float* __restrict__ confneg,
                          int* __restrict__ npos,
                          float* __restrict__ acc) {
  int idx = blockIdx.x * 256 + threadIdx.x;   // exact grid, all lanes active
  int b = idx / P_;
  int p = idx - b * P_;

  float ov = overlap[idx];
  bool pos = ov >= THRESH;

  float2 sc = scores2[idx];
  float mx = fmaxf(sc.x, sc.y);
  float lse = mx + logf(expf(sc.x - mx) + expf(sc.y - mx));
  float conf = lse - (pos ? sc.y : sc.x);     // -log_softmax[label]
  conf = fmaxf(conf, 0.0f);

  float locp = 0.0f;
  if (pos) {
    int obj = objbest[idx];
    float4 bx = boxes4[b * M_ + obj];
    float4 pr = priors4[p];
    float cx = 0.5f * (bx.x + bx.z), cy = 0.5f * (bx.y + bx.w);
    float w = bx.z - bx.x, h = bx.w - bx.y;
    float g0 = (cx - pr.x) / (pr.z * 0.1f);
    float g1 = (cy - pr.y) / (pr.w * 0.1f);
    float g2 = logf(w / pr.z) * 5.0f;
    float g3 = logf(h / pr.w) * 5.0f;
    float4 pl = locs4[idx];
    locp = fabsf(pl.x - g0) + fabsf(pl.y - g1) + fabsf(pl.z - g2) + fabsf(pl.w - g3);
    atomicAdd(&npos[b], 1);
  }
  confneg[idx] = pos ? 0.0f : conf;

  // Reconverged: EXEC all ones -> legal WMMA.
  float confp    = pos ? conf : 0.0f;
  float wsum_loc  = wave_sum_wmma(locp);
  float wsum_conf = wave_sum_wmma(confp);
  if ((threadIdx.x & 31) == 0) {
    atomicAdd(&acc[0], wsum_conf);  // conf_pos_sum
    atomicAdd(&acc[1], wsum_loc);   // loc_abs sum over positives
  }
}

// K5: per-image sum of top-(3*n_pos) negatives via in-register radix select --
__global__ void __launch_bounds__(1024) k5_topk(const float* __restrict__ confneg,
                                                const int* __restrict__ npos,
                                                float* __restrict__ acc) {
  const int IT = (P_ + 1023) / 1024;          // 24 elements per thread
  int b = blockIdx.x;
  int tid = threadIdx.x;
  unsigned u[IT];
#pragma unroll
  for (int i = 0; i < IT; ++i) {
    int p = tid + i * 1024;
    // values are >= 0 so uint bit order == float order; pad with 0 (never
    // counted: every candidate threshold is >= 1 as a bit pattern)
    u[i] = (p < P_) ? __float_as_uint(confneg[(size_t)b * P_ + p]) : 0u;
  }
  int k = RATIO * npos[b];                    // uniform across block
  if (k <= 0) return;

  __shared__ int   cnt_s;
  __shared__ float sum_s;

  // t = k-th largest value (greedy bit descent: max v with count(u>=v) >= k)
  unsigned prefix = 0;
  for (int bit = 31; bit >= 0; --bit) {
    unsigned cand = prefix | (1u << bit);
    int local = 0;
#pragma unroll
    for (int i = 0; i < IT; ++i) local += (u[i] >= cand) ? 1 : 0;
    __syncthreads();
    if (tid == 0) cnt_s = 0;
    __syncthreads();
    if (local) atomicAdd(&cnt_s, local);
    __syncthreads();
    if (cnt_s >= k) prefix = cand;
  }
  float t = __uint_as_float(prefix);

  // top-k sum = sum(v > t) + (k - count(v > t)) * t  (exact under ties)
  float lsum = 0.0f;
  int lgt = 0;
#pragma unroll
  for (int i = 0; i < IT; ++i) {
    if (u[i] > prefix) { lsum += __uint_as_float(u[i]); ++lgt; }
  }
  __syncthreads();
  if (tid == 0) { cnt_s = 0; sum_s = 0.0f; }
  __syncthreads();
  if (lgt) atomicAdd(&cnt_s, lgt);
  if (lsum != 0.0f) atomicAdd(&sum_s, lsum);
  __syncthreads();
  if (tid == 0) {
    float S = sum_s + (float)(k - cnt_s) * t;
    atomicAdd(&acc[2], S);                    // conf_hard_sum
  }
}

// K6: final scalar -----------------------------------------------------------
__global__ void k6_final(const int* __restrict__ npos,
                         const float* __restrict__ acc,
                         float* __restrict__ out) {
  if (threadIdx.x == 0 && blockIdx.x == 0) {
    int tp = 0;
    for (int b = 0; b < B_; ++b) tp += npos[b];
    float tpf = (float)tp;
    float conf_loss = (acc[2] + acc[0]) / tpf;
    float loc_loss = acc[1] / (tpf * 4.0f);
    out[0] = conf_loss + loc_loss;
  }
}

extern "C" void kernel_launch(void* const* d_in, const int* in_sizes, int n_in,
                              void* d_out, int out_size, void* d_ws, size_t ws_size,
                              hipStream_t stream) {
  const float4* locs4   = (const float4*)d_in[0];   // (B,P,4) f32
  const float2* scores2 = (const float2*)d_in[1];   // (B,P,2) f32
  const float4* boxes4  = (const float4*)d_in[2];   // (B,M,4) f32
  const float4* priors4 = (const float4*)d_in[3];   // (P,4)   f32
  float* out = (float*)d_out;

  const size_t BP = (size_t)B_ * P_;
  float* overlap = (float*)d_ws;                    // BP floats
  int*   objbest = (int*)(overlap + BP);            // BP ints
  float* confneg = (float*)(objbest + BP);          // BP floats
  int*   priorfo = (int*)(confneg + BP);            // B*M ints
  int*   npos    = priorfo + B_ * M_;               // B ints
  float* acc     = (float*)(npos + B_);             // [conf_pos, loc, conf_hard]

  const int nb = (int)(BP / 256);                   // 6141, exact

  k0_zero<<<1, 64, 0, stream>>>(npos, acc);
  k1_best_obj<<<nb, 256, 0, stream>>>(boxes4, priors4, overlap, objbest);
  k2_best_prior<<<B_ * M_, 256, 0, stream>>>(boxes4, priors4, priorfo);
  k3_force<<<B_, 32, 0, stream>>>(priorfo, overlap, objbest);
  k4_losses<<<nb, 256, 0, stream>>>(locs4, scores2, boxes4, priors4,
                                    overlap, objbest, confneg, npos, acc);
  k5_topk<<<B_, 1024, 0, stream>>>(confneg, npos, acc);
  k6_final<<<1, 32, 0, stream>>>(npos, acc, out);
}
